// PlanetPolicyModel_90975997263899
// MI455X (gfx1250) — matrix operand order repaired and checked by reading, not compiled, not run on hardware
//
#include <hip/hip_runtime.h>
#include <hip/hip_bf16.h>

// ---------------- model dims ----------------
#define BATCH 1024
#define P     50
#define FLN   200
#define DP    24
#define DF    16
#define DG    16
#define E     192
#define FDIM  128
#define G     384
#define NH    8
#define NL    4
#define FFN   768
#define NB    8
#define CD    128
#define CH    4
#define COND  576            // E+G
#define BP    (BATCH*P)      // 51200
#define BF    (BATCH*FLN)    // 204800

typedef __attribute__((ext_vector_type(16))) _Float16 v16h;
typedef __attribute__((ext_vector_type(2)))  _Float16 h2;
typedef __attribute__((ext_vector_type(8)))  float    v8f;

__device__ __forceinline__ float geluf(float x){ return 0.5f*x*(1.0f+erff(x*0.70710678118654752f)); }
__device__ __forceinline__ float sigmf(float x){ return 1.0f/(1.0f+__expf(-x)); }
__device__ __forceinline__ v16h zero16h(){
  v16h v;
  for (int i = 0; i < 16; ++i) v[i] = (_Float16)0.0f;
  return v;
}
__device__ __forceinline__ v8f zero8f(){
  v8f v;
  for (int i = 0; i < 8; ++i) v[i] = 0.0f;
  return v;
}
// load one 16x32 f16 A-layout fragment from a K-major row base
__device__ __forceinline__ v16h loadA(const _Float16* __restrict__ base, int k0, int kh){
  v16h a;
  #pragma unroll
  for (int p = 0; p < 8; ++p) {
    int ka = k0 + ((p & 4) << 2) + kh*8 + (p & 3)*2;
    h2 v = *(const h2*)(base + ka);
    a[2*p] = v[0]; a[2*p+1] = v[1];
  }
  return a;
}
// load one 32x16 f16 B-layout fragment from a K-major row base
__device__ __forceinline__ v16h loadB(const _Float16* __restrict__ base, int k0, int kh){
  v16h b;
  #pragma unroll
  for (int p = 0; p < 8; ++p) {
    int kb = k0 + ((p & 4) << 2) + kh*8 + (p & 3)*2;
    h2 v = *(const h2*)(base + kb);
    b[2*p] = v[0]; b[2*p+1] = v[1];
  }
  return b;
}

// ============================================================
// Generic NT GEMM:  C[M,N] (+)= A[M,K] * W[N,K]^T
// f16 inputs, f32 accumulate via v_wmma_f32_16x16x32_f16.
// M%32==0, N%32==0, K%32==0. 4 waves/block; each wave owns a
// 32x32 C block (2x2 register tiling: 4 WMMAs per k-step with
// A/B fragment reuse -> 2x arithmetic intensity vs 1 tile/wave).
// ============================================================
__global__ void gemm_nt_f16(const _Float16* __restrict__ A, const _Float16* __restrict__ Wt,
                            float* __restrict__ C, int M, int N, int K, int accum)
{
  const int lane = threadIdx.x & 31;
  const int wave = threadIdx.x >> 5;
  const int m0 = blockIdx.x * 32;
  const int n0 = (blockIdx.y * 4 + wave) * 32;
  if (n0 >= N) return;                       // uniform per wave
  const int kh = lane >> 4;
  const int l15 = lane & 15;
  const _Float16* Ap0 = A  + (size_t)(m0 + l15) * K;
  const _Float16* Ap1 = Ap0 + (size_t)16 * K;
  const _Float16* Wp0 = Wt + (size_t)(n0 + l15) * K;
  const _Float16* Wp1 = Wp0 + (size_t)16 * K;
  v8f c00, c01, c10, c11;
  if (accum) {
    #pragma unroll
    for (int r = 0; r < 8; ++r) {
      size_t row0 = (size_t)(m0 + kh*8 + r) * N;
      size_t row1 = row0 + (size_t)16 * N;
      c00[r] = C[row0 + n0 + l15];      c01[r] = C[row0 + n0 + 16 + l15];
      c10[r] = C[row1 + n0 + l15];      c11[r] = C[row1 + n0 + 16 + l15];
    }
  } else { c00 = zero8f(); c01 = zero8f(); c10 = zero8f(); c11 = zero8f(); }
  for (int k0 = 0; k0 < K; k0 += 32) {
    // prefetch 4 k-steps ahead (speculative; dropped if OOB)
    __builtin_prefetch(Ap0 + k0 + 128, 0, 1);
    __builtin_prefetch(Ap1 + k0 + 128, 0, 1);
    __builtin_prefetch(Wp0 + k0 + 128, 0, 1);
    __builtin_prefetch(Wp1 + k0 + 128, 0, 1);
    v16h a0 = loadA(Ap0, k0, kh);
    v16h a1 = loadA(Ap1, k0, kh);
    v16h b0 = loadB(Wp0, k0, kh);
    v16h b1 = loadB(Wp1, k0, kh);
    c00 = __builtin_amdgcn_wmma_f32_16x16x32_f16(false, a0, false, b0, (short)0, c00, false, false);
    c01 = __builtin_amdgcn_wmma_f32_16x16x32_f16(false, a0, false, b1, (short)0, c01, false, false);
    c10 = __builtin_amdgcn_wmma_f32_16x16x32_f16(false, a1, false, b0, (short)0, c10, false, false);
    c11 = __builtin_amdgcn_wmma_f32_16x16x32_f16(false, a1, false, b1, (short)0, c11, false, false);
  }
  #pragma unroll
  for (int r = 0; r < 8; ++r) {
    size_t row0 = (size_t)(m0 + kh*8 + r) * N;
    size_t row1 = row0 + (size_t)16 * N;
    C[row0 + n0 + l15]      = c00[r];
    C[row0 + n0 + 16 + l15] = c01[r];
    C[row1 + n0 + l15]      = c10[r];
    C[row1 + n0 + 16 + l15] = c11[r];
  }
}

// ============================================================
// Elementwise / conversion helpers
// ============================================================
__global__ void k_cvt16(const float* __restrict__ s, _Float16* __restrict__ d, long n){
  long i = (long)blockIdx.x*blockDim.x + threadIdx.x;
  if (i < n) d[i] = (_Float16)s[i];
}
__global__ void k_cvt_pad16(const float* __restrict__ s, _Float16* __restrict__ d,
                            int rows, int K, int Kp){
  long i = (long)blockIdx.x*blockDim.x + threadIdx.x;
  if (i >= (long)rows*Kp) return;
  int r = (int)(i / Kp), k = (int)(i % Kp);
  d[i] = (k < K) ? (_Float16)s[(long)r*K + k] : (_Float16)0.0f;
}
__global__ void k_cvt_slice16(_Float16* __restrict__ d, int dStride, int dOff,
                              const float* __restrict__ s, int sStride, int sOff,
                              int rows, int cols){
  long i = (long)blockIdx.x*blockDim.x + threadIdx.x;
  if (i >= (long)rows*cols) return;
  int r = (int)(i / cols), c = (int)(i % cols);
  d[(long)r*dStride + dOff + c] = (_Float16)s[(long)r*sStride + sOff + c];
}
__global__ void k_bias_act_cvt16(const float* __restrict__ s, const float* __restrict__ bias,
                                 _Float16* __restrict__ d, long rows, int N, int doGelu){
  long i = (long)blockIdx.x*blockDim.x + threadIdx.x;
  if (i >= rows*N) return;
  float x = s[i] + (bias ? bias[i % N] : 0.0f);
  if (doGelu) x = geluf(x);
  d[i] = (_Float16)x;
}
__global__ void k_add_bias(float* __restrict__ C, const float* __restrict__ bias, long rows, int N){
  long i = (long)blockIdx.x*blockDim.x + threadIdx.x;
  if (i < rows*N) C[i] += bias[i % N];
}
// LayerNorm (optionally adds bias to input first); writes f32 and/or f16 output.
__global__ void k_layernorm(const float* __restrict__ s, const float* __restrict__ bias,
                            const float* __restrict__ g, const float* __restrict__ b,
                            float* __restrict__ d32, _Float16* __restrict__ d16,
                            long rows, int N){
  long r = (long)blockIdx.x*blockDim.x + threadIdx.x;
  if (r >= rows) return;
  const float* p = s + r*N;
  float su=0.f, sq=0.f;
  for (int i=0;i<N;++i){ float x = p[i] + (bias?bias[i]:0.f); su+=x; sq+=x*x; }
  float m = su/N, v = sq/N - m*m;
  float inv = rsqrtf(v + 1e-5f);
  for (int i=0;i<N;++i){
    float x = p[i] + (bias?bias[i]:0.f);
    float y = (x-m)*inv*g[i] + b[i];
    if (d32) d32[r*N+i] = y;
    if (d16) d16[r*N+i] = (_Float16)y;
  }
}
__global__ void k_relbias(const float* __restrict__ rel, const float* __restrict__ rw,
                          float* __restrict__ rb){
  long i = (long)blockIdx.x*blockDim.x + threadIdx.x;
  if (i >= (long)BATCH*P*P) return;
  long b = i/(P*P); int pq = (int)(i%(P*P)); int p = pq/P, q = pq%P;
  float t0=rel[i*4+0], t1=rel[i*4+1], t2=rel[i*4+2], t3=rel[i*4+3];
  #pragma unroll
  for (int h=0; h<NH; ++h)
    rb[(((long)b*NH+h)*P + p)*P + q] = rw[h*4]*t0 + rw[h*4+1]*t1 + rw[h*4+2]*t2 + rw[h*4+3]*t3;
}

// ============================================================
// Cross attention (planet->fleet): CH=4 heads, hd=32
// ============================================================
__global__ void k_scores_cross(const _Float16* __restrict__ qc, const _Float16* __restrict__ kc,
                               const unsigned char* __restrict__ fm, _Float16* __restrict__ aH)
{
  int lane = threadIdx.x & 31, wave = threadIdx.x >> 5;
  int mt = blockIdx.x & 3, bh = blockIdx.x >> 2;   // bh = b*CH+h
  int b = bh >> 2, h = bh & 3;
  int nt = blockIdx.y*4 + wave;
  if (nt >= 13) return;
  int kh = lane >> 4;
  int m = mt*16 + (lane & 15);
  int n = nt*16 + (lane & 15);
  v16h a = zero16h(), bb = zero16h();
  const _Float16* qp = qc + ((long)b*P  + m)*CD + h*32;
  const _Float16* kp = kc + ((long)b*FLN + n)*CD + h*32;
  #pragma unroll
  for (int p_=0;p_<8;++p_){
    int ka = ((p_&4)<<2) + kh*8 + (p_&3)*2;
    if (m < P){ h2 v=*(const h2*)(qp+ka); a[2*p_]=v[0]; a[2*p_+1]=v[1]; }
    int kb = ((p_&4)<<2) + kh*8 + (p_&3)*2;
    if (n < FLN){ h2 v=*(const h2*)(kp+kb); bb[2*p_]=v[0]; bb[2*p_+1]=v[1]; }
  }
  v8f acc = zero8f();
  acc = __builtin_amdgcn_wmma_f32_16x16x32_f16(false,a,false,bb,(short)0,acc,false,false);
  #pragma unroll
  for (int r=0;r<8;++r){
    int mm = mt*16 + kh*8 + r;
    if (mm >= P) continue;
    float v = acc[r]*0.17677669529663687f;              // 1/sqrt(32)
    float o = (n < FLN && fm[b*FLN+n]) ? v : -30000.0f;
    aH[(((long)bh)*P + mm)*224 + n] = (_Float16)o;
  }
}
__global__ void k_softmax_cross(_Float16* __restrict__ aH){
  long r = (long)blockIdx.x*blockDim.x + threadIdx.x;
  if (r >= (long)BATCH*CH*P) return;
  _Float16* p = aH + r*224;
  float mx = -1e30f;
  for (int i=0;i<FLN;++i) mx = fmaxf(mx, (float)p[i]);
  float s = 0.f;
  for (int i=0;i<FLN;++i) s += __expf((float)p[i]-mx);
  float inv = 1.0f/s;
  for (int i=0;i<224;++i){
    float v = (i<FLN) ? __expf((float)p[i]-mx)*inv : 0.f;
    p[i] = (_Float16)v;
  }
}
__global__ void k_av_cross(const _Float16* __restrict__ aH, const _Float16* __restrict__ vc,
                           float* __restrict__ co){
  int lane = threadIdx.x & 31, wave = threadIdx.x >> 5;  // block 64: wave = d-tile
  int mt = blockIdx.x & 3, bh = blockIdx.x >> 2;
  int b = bh >> 2, h = bh & 3;
  int kh = lane >> 4;
  int m = mt*16 + (lane & 15);
  int d = wave*16 + (lane & 15);                         // 0..31
  v8f acc = zero8f();
  const _Float16* Ap = aH + ((long)bh*P + m)*224;
  for (int k0=0;k0<224;k0+=32){
    v16h a = zero16h(), bb = zero16h();
    #pragma unroll
    for (int p_=0;p_<8;++p_){
      int ka = k0 + ((p_&4)<<2) + kh*8 + (p_&3)*2;
      if (m < P){ h2 v=*(const h2*)(Ap+ka); a[2*p_]=v[0]; a[2*p_+1]=v[1]; }
      int kb = k0 + ((p_&4)<<2) + kh*8 + (p_&3)*2;
      if (kb   < FLN) bb[2*p_]   = vc[((long)b*FLN+kb  )*CD + h*32 + d];
      if (kb+1 < FLN) bb[2*p_+1] = vc[((long)b*FLN+kb+1)*CD + h*32 + d];
    }
    acc = __builtin_amdgcn_wmma_f32_16x16x32_f16(false,a,false,bb,(short)0,acc,false,false);
  }
  #pragma unroll
  for (int r=0;r<8;++r){
    int mm = mt*16 + kh*8 + r;
    if (mm < P) co[((long)b*P+mm)*CD + h*32 + d] = acc[r];
  }
}

// ============================================================
// Self attention: NH=8 heads, hd=24 (zero padded to K=32)
// ============================================================
__global__ void k_scores_self(const _Float16* __restrict__ qkv, const float* __restrict__ rb,
                              const unsigned char* __restrict__ pm, _Float16* __restrict__ aH)
{
  int lane = threadIdx.x & 31, wave = threadIdx.x >> 5;  // 4 waves = 4 n-tiles
  int mt = blockIdx.x & 3, bh = blockIdx.x >> 2;         // bh = b*NH+h
  int b = bh >> 3, h = bh & 7;
  int kh = lane >> 4;
  int m = mt*16 + (lane & 15);
  int n = wave*16 + (lane & 15);
  v16h a = zero16h(), bb = zero16h();
  const _Float16* qp = qkv + ((long)b*P + m)*576 + h*24;
  const _Float16* kp = qkv + ((long)b*P + n)*576 + 192 + h*24;
  #pragma unroll
  for (int p_=0;p_<8;++p_){
    int ka = ((p_&4)<<2) + kh*8 + (p_&3)*2;
    if (m < P && ka < 24){ h2 v=*(const h2*)(qp+ka); a[2*p_]=v[0]; a[2*p_+1]=v[1]; }
    int kb = ((p_&4)<<2) + kh*8 + (p_&3)*2;
    if (n < P && kb < 24){ h2 v=*(const h2*)(kp+kb); bb[2*p_]=v[0]; bb[2*p_+1]=v[1]; }
  }
  v8f acc = zero8f();
  acc = __builtin_amdgcn_wmma_f32_16x16x32_f16(false,a,false,bb,(short)0,acc,false,false);
  #pragma unroll
  for (int r=0;r<8;++r){
    int mm = mt*16 + kh*8 + r;
    if (mm >= P) continue;
    float o;
    if (n < P){
      float v = acc[r]*0.20412414523193154f;             // 1/sqrt(24)
      v += rb[(((long)b*NH+h)*P + mm)*P + n];
      o = pm[b*P+n] ? v : -30000.0f;
    } else o = -30000.0f;
    aH[(((long)bh)*P + mm)*64 + n] = (_Float16)o;
  }
}
__global__ void k_softmax_self(_Float16* __restrict__ aH){
  long r = (long)blockIdx.x*blockDim.x + threadIdx.x;
  if (r >= (long)BATCH*NH*P) return;
  _Float16* p = aH + r*64;
  float mx = -1e30f;
  for (int i=0;i<P;++i) mx = fmaxf(mx, (float)p[i]);
  float s = 0.f;
  for (int i=0;i<P;++i) s += __expf((float)p[i]-mx);
  float inv = 1.0f/s;
  for (int i=0;i<64;++i){
    float v = (i<P) ? __expf((float)p[i]-mx)*inv : 0.f;
    p[i] = (_Float16)v;
  }
}
__global__ void k_av_self(const _Float16* __restrict__ aH, const _Float16* __restrict__ qkv,
                          float* __restrict__ o){
  int lane = threadIdx.x & 31, wave = threadIdx.x >> 5;  // block 64
  int mt = blockIdx.x & 3, bh = blockIdx.x >> 2;
  int b = bh >> 3, h = bh & 7;
  int kh = lane >> 4;
  int m = mt*16 + (lane & 15);
  int d = wave*16 + (lane & 15);                         // 0..31, store d<24
  v8f acc = zero8f();
  const _Float16* Ap = aH + ((long)bh*P + m)*64;
  for (int k0=0;k0<64;k0+=32){
    v16h a = zero16h(), bb = zero16h();
    #pragma unroll
    for (int p_=0;p_<8;++p_){
      int ka = k0 + ((p_&4)<<2) + kh*8 + (p_&3)*2;
      if (m < P){ h2 v=*(const h2*)(Ap+ka); a[2*p_]=v[0]; a[2*p_+1]=v[1]; }
      int kb = k0 + ((p_&4)<<2) + kh*8 + (p_&3)*2;
      if (kb   < P && d < 24) bb[2*p_]   = qkv[((long)b*P+kb  )*576 + 384 + h*24 + d];
      if (kb+1 < P && d < 24) bb[2*p_+1] = qkv[((long)b*P+kb+1)*576 + 384 + h*24 + d];
    }
    acc = __builtin_amdgcn_wmma_f32_16x16x32_f16(false,a,false,bb,(short)0,acc,false,false);
  }
  #pragma unroll
  for (int r=0;r<8;++r){
    int mm = mt*16 + kh*8 + r;
    if (mm < P && d < 24) o[((long)b*P+mm)*E + h*24 + d] = acc[r];
  }
}

// ============================================================
// Target logits tl[b,p,p] = query . keyp / sqrt(E) with -inf mask
// ============================================================
__global__ void k_tl(const _Float16* __restrict__ qH, const _Float16* __restrict__ kH,
                     const unsigned char* __restrict__ pm, float* __restrict__ tl){
  int lane = threadIdx.x & 31, wave = threadIdx.x >> 5;  // 4 waves = n-tiles
  int mt = blockIdx.x & 3, b = blockIdx.x >> 2;
  int kh = lane >> 4;
  int m = mt*16 + (lane & 15);
  int n = wave*16 + (lane & 15);
  const _Float16* Ap = qH + ((long)b*P + m)*E;
  const _Float16* Bp = kH + ((long)b*P + n)*E;
  v8f acc = zero8f();
  for (int k0=0;k0<E;k0+=32){
    v16h a = zero16h(), bb = zero16h();
    #pragma unroll
    for (int p_=0;p_<8;++p_){
      int ka = k0 + ((p_&4)<<2) + kh*8 + (p_&3)*2;
      if (m < P){ h2 v=*(const h2*)(Ap+ka); a[2*p_]=v[0]; a[2*p_+1]=v[1]; }
      int kb = k0 + ((p_&4)<<2) + kh*8 + (p_&3)*2;
      if (n < P){ h2 v=*(const h2*)(Bp+kb); bb[2*p_]=v[0]; bb[2*p_+1]=v[1]; }
    }
    acc = __builtin_amdgcn_wmma_f32_16x16x32_f16(false,a,false,bb,(short)0,acc,false,false);
  }
  #pragma unroll
  for (int r=0;r<8;++r){
    int mm = mt*16 + kh*8 + r;
    if (mm < P && n < P){
      float v = acc[r]*0.07216878364870323f;             // 1/sqrt(192)
      tl[((long)b*P+mm)*P + n] = pm[b*P+n] ? v : -__builtin_huge_valf();
    }
  }
}

// ============================================================
// Pooling / recurrent / head kernels
// ============================================================
__global__ void k_pool_planet(const float* __restrict__ x, const float* __restrict__ q,
                              const unsigned char* __restrict__ pm, float* __restrict__ pool){
  __shared__ float w[P]; __shared__ float sc[P];
  int b = blockIdx.x, t = threadIdx.x;
  if (t < P){
    float s=0.f; const float* xp = x + ((long)b*P+t)*E;
    for (int e=0;e<E;++e) s += q[e]*xp[e];
    sc[t] = s*0.07216878364870323f;
  }
  __syncthreads();
  if (t == 0){
    float mx=-1e30f; int cnt=0;
    for (int p=0;p<P;++p) if (pm[b*P+p]){ cnt++; mx=fmaxf(mx, sc[p]); }
    if (cnt==0){ for (int p=0;p<P;++p) w[p]=1.0f/P; }
    else {
      float s=0.f;
      for (int p=0;p<P;++p){ float e_=pm[b*P+p]?__expf(sc[p]-mx):0.f; w[p]=e_; s+=e_; }
      float inv=1.f/s; for (int p=0;p<P;++p) w[p]*=inv;
    }
  }
  __syncthreads();
  for (int e=t;e<E;e+=blockDim.x){
    float s=0.f;
    for (int p=0;p<P;++p) s += w[p]*x[((long)b*P+p)*E+e];
    pool[(long)b*E+e] = s;
  }
}
__global__ void k_pool_fleet(const _Float16* __restrict__ x, const float* __restrict__ q,
                             const unsigned char* __restrict__ fm, float* __restrict__ pool){
  __shared__ float w[FLN]; __shared__ float sc[FLN];
  int b = blockIdx.x, t = threadIdx.x;
  if (t < FLN){
    float s=0.f; const _Float16* xp = x + ((long)b*FLN+t)*FDIM;
    for (int e=0;e<FDIM;++e) s += q[e]*(float)xp[e];
    sc[t] = s*0.08838834764831845f;                      // 1/sqrt(128)
  }
  __syncthreads();
  if (t == 0){
    float mx=-1e30f; int cnt=0;
    for (int f=0;f<FLN;++f) if (fm[b*FLN+f]){ cnt++; mx=fmaxf(mx, sc[f]); }
    if (cnt==0){ for (int f=0;f<FLN;++f) w[f]=1.0f/FLN; }
    else {
      float s=0.f;
      for (int f=0;f<FLN;++f){ float e_=fm[b*FLN+f]?__expf(sc[f]-mx):0.f; w[f]=e_; s+=e_; }
      float inv=1.f/s; for (int f=0;f<FLN;++f) w[f]*=inv;
    }
  }
  __syncthreads();
  for (int e=t;e<FDIM;e+=blockDim.x){
    float s=0.f;
    for (int f=0;f<FLN;++f) s += w[f]*(float)x[((long)b*FLN+f)*FDIM+e];
    pool[(long)b*FDIM+e] = s;
  }
}
__global__ void k_grin(const float* __restrict__ pp, const float* __restrict__ fp,
                       const float* __restrict__ gf, _Float16* __restrict__ d){
  long i = (long)blockIdx.x*blockDim.x + threadIdx.x;
  if (i >= (long)BATCH*352) return;
  int c = (int)(i % 352); long b = i / 352;
  float v;
  if (c < E) v = pp[b*E+c];
  else if (c < E+FDIM) v = fp[b*FDIM + c - E];
  else if (c < E+FDIM+DG) v = gf[b*DG + c - E - FDIM];
  else v = 0.f;
  d[i] = (_Float16)v;
}
__global__ void k_lstm(const float* __restrict__ gates, const float* __restrict__ bih,
                       const float* __restrict__ bhh, const float* __restrict__ c0,
                       float* __restrict__ hO, float* __restrict__ cO){
  long i = (long)blockIdx.x*blockDim.x + threadIdx.x;
  if (i >= (long)BATCH*G) return;
  long b = i / G; int g = (int)(i % G);
  const float* gp = gates + b*4*G;
  float ig = sigmf(gp[g]       + bih[g]       + bhh[g]);
  float fg = sigmf(gp[G+g]     + bih[G+g]     + bhh[G+g]);
  float gg = tanhf(gp[2*G+g]   + bih[2*G+g]   + bhh[2*G+g]);
  float og = sigmf(gp[3*G+g]   + bih[3*G+g]   + bhh[3*G+g]);
  float c = fg*c0[i] + ig*gg;
  cO[i] = c; hO[i] = og*tanhf(c);
}
__global__ void k_atl(const float* __restrict__ x, const float* __restrict__ h,
                      const float* __restrict__ aw, const float* __restrict__ ab,
                      float* __restrict__ atl, int* __restrict__ idx){
  long i = (long)blockIdx.x*blockDim.x + threadIdx.x;
  if (i >= (long)BP) return;
  long b = i / P;
  const float* xp = x + i*E;
  const float* hp = h + b*G;
  float best = -1e30f; int bi = 0; float vals[3];
  for (int j=0;j<3;++j){
    const float* w = aw + (long)j*COND;
    float s = ab[j];
    for (int e=0;e<E;++e) s += xp[e]*w[e];
    for (int g=0;g<G;++g) s += hp[g]*w[E+g];
    vals[j] = s; atl[i*3+j] = s;
  }
  for (int j=0;j<3;++j) if (vals[j] > best){ best = vals[j]; bi = j; }
  idx[i] = bi;
}
__global__ void k_qlst(const float* __restrict__ h, const float* __restrict__ emb,
                       const float* __restrict__ wq, float* __restrict__ out){
  long i = (long)blockIdx.x*blockDim.x + threadIdx.x;
  if (i >= (long)BATCH*3*E) return;
  int e = (int)(i % E); long bj = i / E; int j = (int)(bj % 3); long b = bj / 3;
  const float* w = wq + (long)e*COND;
  float s = 0.f;
  for (int k=0;k<E;++k) s += emb[j*COND+k]*w[k];
  for (int g=0;g<G;++g) s += (h[b*G+g]+emb[j*COND+E+g])*w[E+g];
  out[i] = s;
}
__global__ void k_query(const float* __restrict__ qE, const float* __restrict__ qlst,
                        const int* __restrict__ idx, _Float16* __restrict__ qH){
  long i = (long)blockIdx.x*blockDim.x + threadIdx.x;
  if (i >= (long)BP*E) return;
  int e = (int)(i % E); long bp = i / E; long b = bp / P;
  qH[i] = (_Float16)(qE[i] + qlst[(b*3 + idx[bp])*(long)E + e]);
}
__global__ void k_tgt(const float* __restrict__ tl, const float* __restrict__ rel,
                      int* __restrict__ tgt, float* __restrict__ dist){
  long i = (long)blockIdx.x*blockDim.x + threadIdx.x;
  if (i >= (long)BP) return;
  const float* row = tl + i*P;
  float best = row[0]; int bi = 0;
  for (int n=1;n<P;++n){ float v = row[n]; if (v > best){ best = v; bi = n; } }
  tgt[i] = bi;
  dist[i] = rel[((long)i*P + bi)*4];
}
__global__ void k_gather_ctxt(const float* __restrict__ x, const int* __restrict__ tgt,
                              _Float16* __restrict__ cat){
  long i = (long)blockIdx.x*blockDim.x + threadIdx.x;
  if (i >= (long)BP*E) return;
  int e = (int)(i % E); long bp = i / E; long b = bp / P;
  cat[bp*384 + 192 + e] = (_Float16)x[((long)b*P + tgt[bp])*E + e];
}
__global__ void k_wd(const float* __restrict__ am1w, const float* __restrict__ dw,
                     float* __restrict__ wd){
  int f = blockIdx.x*blockDim.x + threadIdx.x;
  if (f >= FFN) return;
  const float* w = am1w + (long)f*1344 + 768;
  float s = 0.f;
  for (int c=0;c<COND;++c) s += w[c]*dw[c];
  wd[f] = s;
}
__global__ void k_hlst(const float* __restrict__ h, const float* __restrict__ emb,
                       const float* __restrict__ am1w, float* __restrict__ out){
  long i = (long)blockIdx.x*blockDim.x + threadIdx.x;
  if (i >= (long)BATCH*3*FFN) return;
  int f = (int)(i % FFN); long bj = i / FFN; int j = (int)(bj % 3); long b = bj / 3;
  const float* w = am1w + (long)f*1344;
  float s = 0.f;
  for (int e=0;e<E;++e) s += emb[j*COND+e]*w[e];
  for (int g=0;g<G;++g) s += (h[b*G+g]+emb[j*COND+E+g])*w[E+g];
  out[i] = s;
}
__global__ void k_amt_epi(float* __restrict__ amt, const float* __restrict__ hlst,
                          const float* __restrict__ wd, const float* __restrict__ b1,
                          const int* __restrict__ idx, const float* __restrict__ dist){
  long i = (long)blockIdx.x*blockDim.x + threadIdx.x;
  if (i >= (long)BP*FFN) return;
  int f = (int)(i % FFN); long bp = i / FFN; long b = bp / P;
  float v = amt[i] + hlst[(b*3 + idx[bp])*(long)FFN + f] + dist[bp]*wd[f] + b1[f];
  amt[i] = geluf(v);
}
__global__ void k_am2(const float* __restrict__ amt, const float* __restrict__ w,
                      const float* __restrict__ b2, float* __restrict__ out){
  long i = (long)blockIdx.x*blockDim.x + threadIdx.x;
  if (i >= (long)BP*NB) return;
  int j = (int)(i % NB); long bp = i / NB;
  const float* a = amt + bp*FFN; const float* wr = w + (long)j*FFN;
  float s = b2[j];
  for (int f=0;f<FFN;++f) s += a[f]*wr[f];
  out[i] = s;
}
__global__ void k_vhead(const float* __restrict__ h, const float* __restrict__ w1,
                        const float* __restrict__ b1, const float* __restrict__ w2,
                        const float* __restrict__ b2, float* __restrict__ out, int doTanh){
  __shared__ float red[192];
  int b = blockIdx.x, t = threadIdx.x;       // 192 threads
  const float* hp = h + (long)b*G;
  const float* wr = w1 + (long)t*G;
  float s = b1[t];
  for (int g=0;g<G;++g) s += hp[g]*wr[g];
  red[t] = geluf(s)*w2[t];
  __syncthreads();
  for (int st=96; st>=3; st>>=1){ if (t < st) red[t] += red[t+st]; __syncthreads(); }
  if (t == 0){
    float v = red[0]+red[1]+red[2] + b2[0];
    out[b] = doTanh ? tanhf(v) : v;
  }
}

// ============================================================
// Host driver
// ============================================================
static inline void gemmNT(hipStream_t s, const _Float16* A, const _Float16* Wt,
                          float* C, int M, int N, int K, int accum){
  dim3 g(M/32, (N/32 + 3)/4);
  gemm_nt_f16<<<g, 128, 0, s>>>(A, Wt, C, M, N, K, accum);
}
#define LB(n) dim3((unsigned)(((n)+255)/256)), dim3(256)

extern "C" void kernel_launch(void* const* d_in, const int* in_sizes, int n_in,
                              void* d_out, int out_size, void* d_ws, size_t ws_size,
                              hipStream_t stream)
{
  (void)in_sizes; (void)n_in; (void)out_size;
  const float* planet = (const float*)d_in[0];
  const float* fleet  = (const float*)d_in[1];
  const unsigned char* fmask = (const unsigned char*)d_in[2];
  const float* gfeat  = (const float*)d_in[3];
  const unsigned char* pmask = (const unsigned char*)d_in[4];
  const float* rel    = (const float*)d_in[5];
  const float* h0     = (const float*)d_in[6];
  const float* c0     = (const float*)d_in[7];
#define PRM(i) ((const float*)d_in[8+(i)])
  // param leaves assumed in dict insertion order (see _make_params):
  // 0 pe0_w 1 pe0_b 2 pe2_w 3 pe2_b 4 pe_ln_g 5 pe_ln_b
  // 6 fe0_w 7 fe0_b 8 fe2_w 9 fe2_b 10 fe_ln_g 11 fe_ln_b
  // 12 cq_w 13 ck_w 14 cv_w 15 co_w 16 co_b 17 cn_g 18 cn_b 19 rel_w
  // 20+12i: ln1_g ln1_b qkv_w qkv_b o_w o_b ln2_g ln2_b f1_w f1_b f2_w f2_b
  // 68 ppq 69 fpq 70 g0_w 71 g0_b 72 g2_w 73 g2_b 74 g_ln_g 75 g_ln_b
  // 76 wih 77 whh 78 bih 79 bhh 80 ath_w 81 ath_b 82 wq_w 83 wk_w
  // 84 at_emb 85 dist_w 86 am1_w 87 am1_b 88 am2_w 89 am2_b
  // 90-93 vo, 94-97 vs, 98-101 vh

  // ---- workspace carve ----
  char* wsb = (char*)d_ws;
  size_t off = 0;
  auto take = [&](size_t bytes)->char*{ char* p = wsb + off; off = (off + bytes + 255) & ~(size_t)255; return p; };
  float*    xbuf = (float*)   take((size_t)BP*E*4);
  _Float16* peH  = (_Float16*)take((size_t)BP*E*2);
  _Float16* feH  = (_Float16*)take((size_t)BF*FDIM*2);
  float*    rb   = (float*)   take((size_t)BATCH*NH*P*P*4);
  float*    bigB = (float*)   take((size_t)BP*FFN*4);          // 157 MB, reused heavily
  char*     fh1  = take(91750400);                             // biggest f16 pool
  char*     fh2  = take(65536000);
  char*     fh3  = take(58982400);
  float*    ppool= (float*)   take((size_t)BATCH*E*4);
  float*    fpool= (float*)   take((size_t)BATCH*FDIM*4);
  _Float16* grinH= (_Float16*)take((size_t)BATCH*352*2);
  float*    qlst = (float*)   take((size_t)BATCH*3*E*4);
  float*    hlst = (float*)   take((size_t)BATCH*3*FFN*4);
  float*    wd   = (float*)   take(FFN*4);
  int*      atIdx= (int*)     take((size_t)BP*4);
  int*      tgtIdx=(int*)     take((size_t)BP*4);
  float*    dist = (float*)   take((size_t)BP*4);
  _Float16* wbase= (_Float16*)take(8u*1024u*1024u);
  if (off > ws_size) return;   // not enough scratch; bail safely

  size_t wo = 0;
  auto wtake = [&](size_t halfs)->_Float16*{ _Float16* p = wbase + wo; wo += (halfs + 63) & ~(size_t)63; return p; };
  _Float16* w_pe0 = wtake(192*32);
  _Float16* w_pe2 = wtake(192*192);
  _Float16* w_fe0 = wtake(128*32);
  _Float16* w_fe2 = wtake(128*128);
  _Float16* w_cq  = wtake(128*192);
  _Float16* w_ck  = wtake(128*128);
  _Float16* w_cv  = wtake(128*128);
  _Float16* w_co  = wtake(192*320);
  _Float16 *w_qkv[NL], *w_o[NL], *w_f1[NL], *w_f2[NL];
  for (int i=0;i<NL;++i){ w_qkv[i]=wtake(576*192); w_o[i]=wtake(192*192); w_f1[i]=wtake(768*192); w_f2[i]=wtake(192*768); }
  _Float16* w_g0  = wtake(384*352);
  _Float16* w_g2  = wtake(384*384);
  _Float16* w_wih = wtake(1536*384);
  _Float16* w_whh = wtake(1536*384);
  _Float16* w_wqE = wtake(192*192);
  _Float16* w_wk  = wtake(192*192);
  _Float16* w_am1 = wtake(768*384);

  auto cvtpad = [&](const float* s, _Float16* d, int rows, int K, int Kp){
    long n = (long)rows*Kp;
    k_cvt_pad16<<<LB(n), 0, stream>>>(s, d, rows, K, Kp);
  };
  // ---- weight conversions (f32 -> f16, K padded to %32) ----
  cvtpad(PRM(0),  w_pe0, 192, 24, 32);
  cvtpad(PRM(2),  w_pe2, 192, 192, 192);
  cvtpad(PRM(6),  w_fe0, 128, 16, 32);
  cvtpad(PRM(8),  w_fe2, 128, 128, 128);
  cvtpad(PRM(12), w_cq,  128, 192, 192);
  cvtpad(PRM(13), w_ck,  128, 128, 128);
  cvtpad(PRM(14), w_cv,  128, 128, 128);
  cvtpad(PRM(15), w_co,  192, 320, 320);
  for (int i=0;i<NL;++i){
    int pb = 20 + i*12;
    cvtpad(PRM(pb+2), w_qkv[i], 576, 192, 192);
    cvtpad(PRM(pb+4), w_o[i],   192, 192, 192);
    cvtpad(PRM(pb+8), w_f1[i],  768, 192, 192);
    cvtpad(PRM(pb+10),w_f2[i],  192, 768, 768);
  }
  cvtpad(PRM(70), w_g0, 384, 336, 352);
  cvtpad(PRM(72), w_g2, 384, 384, 384);
  cvtpad(PRM(76), w_wih, 1536, 384, 384);
  cvtpad(PRM(77), w_whh, 1536, 384, 384);
  k_cvt_slice16<<<LB((long)192*192), 0, stream>>>(w_wqE, 192, 0,   PRM(82), COND, 0,   192, 192);
  cvtpad(PRM(83), w_wk, 192, 192, 192);
  k_cvt_slice16<<<LB((long)768*192), 0, stream>>>(w_am1, 384, 0,   PRM(86), 1344, 0,   768, 192);
  k_cvt_slice16<<<LB((long)768*192), 0, stream>>>(w_am1, 384, 192, PRM(86), 1344, 576, 768, 192);

  // ---- output segments ----
  float* out   = (float*)d_out;
  float* o_atl = out;
  float* o_tl  = out + 153600;
  float* o_amt = out + 2713600;
  float* o_vo  = out + 3123200;
  float* o_vs  = out + 3124224;
  float* o_vh  = out + 3125248;
  float* o_h   = out + 3126272;
  float* o_c   = out + 3519488;

  // ---- planet encoder ----
  _Float16* pfH = (_Float16*)fh1;
  cvtpad(planet, pfH, BP, DP, 32);
  gemmNT(stream, pfH, w_pe0, bigB, BP, E, 32, 0);
  _Float16* t16 = (_Float16*)fh3;
  k_bias_act_cvt16<<<LB((long)BP*E), 0, stream>>>(bigB, PRM(1), t16, (long)BP, E, 1);
  gemmNT(stream, t16, w_pe2, bigB, BP, E, E, 0);
  k_layernorm<<<LB(BP), 0, stream>>>(bigB, PRM(3), PRM(4), PRM(5), xbuf, ((_Float16*)0), (long)BP, E);
  k_cvt16<<<LB((long)BP*E), 0, stream>>>(xbuf, peH, (long)BP*E);

  // ---- fleet encoder ----
  _Float16* ffH = (_Float16*)fh2;
  cvtpad(fleet, ffH, BF, DF, 32);
  gemmNT(stream, ffH, w_fe0, bigB, BF, FDIM, 32, 0);
  _Float16* fm16 = (_Float16*)fh3;
  k_bias_act_cvt16<<<LB((long)BF*FDIM), 0, stream>>>(bigB, PRM(7), fm16, (long)BF, FDIM, 1);
  gemmNT(stream, fm16, w_fe2, bigB, BF, FDIM, FDIM, 0);
  k_layernorm<<<LB(BF), 0, stream>>>(bigB, PRM(9), PRM(10), PRM(11), ((float*)0), feH, (long)BF, FDIM);

  // ---- relational bias ----
  k_relbias<<<LB((long)BATCH*P*P), 0, stream>>>(rel, PRM(19), rb);

  // ---- cross attention ----
  _Float16* kcH = (_Float16*)fh3;
  gemmNT(stream, feH, w_ck, bigB, BF, CD, FDIM, 0);
  k_cvt16<<<LB((long)BF*CD), 0, stream>>>(bigB, kcH, (long)BF*CD);
  _Float16* vcH = (_Float16*)fh2;
  gemmNT(stream, feH, w_cv, bigB, BF, CD, FDIM, 0);
  k_cvt16<<<LB((long)BF*CD), 0, stream>>>(bigB, vcH, (long)BF*CD);
  _Float16* qcH = (_Float16*)(fh2 + (size_t)BF*CD*2);
  gemmNT(stream, peH, w_cq, bigB, BP, CD, E, 0);
  k_cvt16<<<LB((long)BP*CD), 0, stream>>>(bigB, qcH, (long)BP*CD);
  _Float16* aHc = (_Float16*)fh1;
  k_scores_cross<<<dim3(BATCH*CH*4, 4), 128, 0, stream>>>(qcH, kcH, fmask, aHc);
  k_softmax_cross<<<LB((long)BATCH*CH*P), 0, stream>>>(aHc);
  float* co = bigB;
  k_av_cross<<<BATCH*CH*4, 64, 0, stream>>>(aHc, vcH, co);
  _Float16* catH = (_Float16*)fh1;            // aHc dead
  k_cvt_slice16<<<LB((long)BP*E), 0, stream>>>(catH, 320, 0,   xbuf, E,  0, BP, E);
  k_cvt_slice16<<<LB((long)BP*CD), 0, stream>>>(catH, 320, 192, co,   CD, 0, BP, CD);
  gemmNT(stream, catH, w_co, bigB, BP, E, 320, 0);
  k_layernorm<<<LB(BP), 0, stream>>>(bigB, PRM(16), PRM(17), PRM(18), xbuf, ((_Float16*)0), (long)BP, E);

  // ---- transformer layers ----
  for (int L=0; L<NL; ++L){
    int pb = 20 + L*12;
    _Float16* xlnH = (_Float16*)fh3;
    k_layernorm<<<LB(BP), 0, stream>>>(xbuf, ((const float*)0), PRM(pb+0), PRM(pb+1),
                                       ((float*)0), xlnH, (long)BP, E);
    gemmNT(stream, xlnH, w_qkv[L], bigB, BP, 3*E, E, 0);
    _Float16* qkvH = (_Float16*)fh1;
    k_bias_act_cvt16<<<LB((long)BP*3*E), 0, stream>>>(bigB, PRM(pb+3), qkvH, (long)BP, 3*E, 0);
    _Float16* aH2 = (_Float16*)fh2;
    k_scores_self<<<BATCH*NH*4, 128, 0, stream>>>(qkvH, rb, pmask, aH2);
    k_softmax_self<<<LB((long)BATCH*NH*P), 0, stream>>>(aH2);
    k_av_self<<<BATCH*NH*4, 64, 0, stream>>>(aH2, qkvH, bigB);
    _Float16* oH = (_Float16*)fh3;
    k_cvt16<<<LB((long)BP*E), 0, stream>>>(bigB, oH, (long)BP*E);
    gemmNT(stream, oH, w_o[L], xbuf, BP, E, E, 1);          // residual accumulate
    k_add_bias<<<LB((long)BP*E), 0, stream>>>(xbuf, PRM(pb+5), (long)BP, E);
    _Float16* ln2H = (_Float16*)fh3;
    k_layernorm<<<LB(BP), 0, stream>>>(xbuf, ((const float*)0), PRM(pb+6), PRM(pb+7),
                                       ((float*)0), ln2H, (long)BP, E);
    gemmNT(stream, ln2H, w_f1[L], bigB, BP, FFN, E, 0);
    _Float16* ffnH = (_Float16*)fh1;
    k_bias_act_cvt16<<<LB((long)BP*FFN), 0, stream>>>(bigB, PRM(pb+9), ffnH, (long)BP, FFN, 1);
    gemmNT(stream, ffnH, w_f2[L], xbuf, BP, E, FFN, 1);     // residual accumulate
    k_add_bias<<<LB((long)BP*E), 0, stream>>>(xbuf, PRM(pb+11), (long)BP, E);
  }

  // ---- pooling + global encoder + LSTM ----
  k_pool_planet<<<BATCH, 256, 0, stream>>>(xbuf, PRM(68), pmask, ppool);
  k_pool_fleet<<<BATCH, 256, 0, stream>>>(feH, PRM(69), fmask, fpool);
  k_grin<<<LB((long)BATCH*352), 0, stream>>>(ppool, fpool, gfeat, grinH);
  gemmNT(stream, grinH, w_g0, bigB, BATCH, G, 352, 0);
  _Float16* g1H = (_Float16*)fh3;
  k_bias_act_cvt16<<<LB((long)BATCH*G), 0, stream>>>(bigB, PRM(71), g1H, (long)BATCH, G, 1);
  gemmNT(stream, g1H, w_g2, bigB, BATCH, G, G, 0);
  _Float16* grH = (_Float16*)(fh3 + 1048576);
  k_layernorm<<<LB(BATCH), 0, stream>>>(bigB, PRM(73), PRM(74), PRM(75), ((float*)0), grH, (long)BATCH, G);
  _Float16* h0H = (_Float16*)(fh3 + 2097152);
  k_cvt16<<<LB((long)BATCH*G), 0, stream>>>(h0, h0H, (long)BATCH*G);
  float* gates = bigB;
  gemmNT(stream, grH, w_wih, gates, BATCH, 4*G, G, 0);
  gemmNT(stream, h0H, w_whh, gates, BATCH, 4*G, G, 1);
  k_lstm<<<LB((long)BATCH*G), 0, stream>>>(gates, PRM(78), PRM(79), c0, o_h, o_c);

  // ---- action-type head + target head ----
  k_atl<<<LB((long)BP), 0, stream>>>(xbuf, o_h, PRM(80), PRM(81), o_atl, atIdx);
  _Float16* ctxH = (_Float16*)fh3;
  k_cvt16<<<LB((long)BP*E), 0, stream>>>(xbuf, ctxH, (long)BP*E);
  gemmNT(stream, ctxH, w_wk, bigB, BP, E, E, 0);
  _Float16* keypH = (_Float16*)(fh3 + (size_t)BP*E*2);
  k_cvt16<<<LB((long)BP*E), 0, stream>>>(bigB, keypH, (long)BP*E);
  gemmNT(stream, ctxH, w_wqE, bigB, BP, E, E, 0);           // planet-ctx part of query
  k_qlst<<<LB((long)BATCH*3*E), 0, stream>>>(o_h, PRM(84), PRM(82), qlst);
  _Float16* queryH = (_Float16*)(fh3 + (size_t)BP*E*4);
  k_query<<<LB((long)BP*E), 0, stream>>>(bigB, qlst, atIdx, queryH);
  k_tl<<<BATCH*4, 128, 0, stream>>>(queryH, keypH, pmask, o_tl);
  k_tgt<<<LB((long)BP), 0, stream>>>(o_tl, rel, tgtIdx, dist);

  // ---- amount head (dist_emb folded as rank-1, lstm part as 3-entry table) ----
  _Float16* catH2 = (_Float16*)fh1;
  k_cvt_slice16<<<LB((long)BP*E), 0, stream>>>(catH2, 384, 0, xbuf, E, 0, BP, E);
  k_gather_ctxt<<<LB((long)BP*E), 0, stream>>>(xbuf, tgtIdx, catH2);
  k_wd<<<LB(FFN), 0, stream>>>(PRM(86), PRM(85), wd);
  k_hlst<<<LB((long)BATCH*3*FFN), 0, stream>>>(o_h, PRM(84), PRM(86), hlst);
  gemmNT(stream, catH2, w_am1, bigB, BP, FFN, 384, 0);
  k_amt_epi<<<LB((long)BP*FFN), 0, stream>>>(bigB, hlst, wd, PRM(87), atIdx, dist);
  k_am2<<<LB((long)BP*NB), 0, stream>>>(bigB, PRM(88), PRM(89), o_amt);

  // ---- value heads ----
  k_vhead<<<BATCH, 192, 0, stream>>>(o_h, PRM(90), PRM(91), PRM(92), PRM(93), o_vo, 1);
  k_vhead<<<BATCH, 192, 0, stream>>>(o_h, PRM(94), PRM(95), PRM(96), PRM(97), o_vs, 0);
  k_vhead<<<BATCH, 192, 0, stream>>>(o_h, PRM(98), PRM(99), PRM(100), PRM(101), o_vh, 0);
#undef PRM
}
